// TensorBase_30142080483421
// MI455X (gfx1250) — matrix operand order repaired and checked by reading, not compiled, not run on hardware
//
#include <hip/hip_runtime.h>
#include <hip/hip_bf16.h>
#include <math.h>

typedef __attribute__((ext_vector_type(16))) _Float16 v16h;
typedef __attribute__((ext_vector_type(8)))  float    v8f;
typedef __attribute__((ext_vector_type(4)))  unsigned int v4u;

union FragU { v4u u[2]; v16h h; };

#define N_RAYS 4096
#define NS     128
#define APPD   27
#define KP1    416      // 390 padded to 13*32
#define FC     128
#define STRA   424      // full-A LDS row stride (halves): 848B = 53*16, conflict-skewed
#define STRH   136      // H LDS row stride (halves): 272B = 17*16

// f16 weight workspace layout (in halves)
#define W1H_OFF 0
#define W2H_OFF (128*416)
#define W3H_OFF (128*416 + 128*128)
#define WTOT    (128*416 + 128*128 + 16*128)

// LDS layout (bytes)
#define FEATA_OFF 0                 // 128*424 halves = 108544 B (live: build + GEMM1)
#define H1_OFF    0                 // overlay after GEMM1 barrier: 34816 B
#define H2_OFF    34816             // overlay: 34816 B
#define WRAY_OFF  108544            // 128 f32
#define RED_OFF   109056            // 128*4 f32
#define SMEM_TOT  111104

// ---- pre-pass: weights -> f16, transposed to B-fragment [col][k] layout ----
__global__ void prep_weights(const float* __restrict__ W1,
                             const float* __restrict__ W2,
                             const float* __restrict__ W3,
                             _Float16* __restrict__ ws) {
  int i = blockIdx.x * 256 + threadIdx.x;
  if (i < W2H_OFF) {                       // W1h[n*416 + k] = W1[k*128 + n]
    int n = i / KP1, k = i - n * KP1;
    ws[i] = (k < 390) ? (_Float16)W1[k * 128 + n] : (_Float16)0.0f;
  } else if (i < W3H_OFF) {                // W2h[n*128 + k] = W2[k*128 + n]
    int j = i - W2H_OFF;
    int n = j / FC, k = j - n * FC;
    ws[i] = (_Float16)W2[k * 128 + n];
  } else if (i < WTOT) {                   // W3h[n*128 + k] = W3[k*3 + n], n<3 else 0
    int j = i - W3H_OFF;
    int n = j / FC, k = j - n * FC;
    ws[i] = (n < 3) ? (_Float16)W3[k * 3 + n] : (_Float16)0.0f;
  }
}

__device__ __forceinline__ v16h load_frag(const _Float16* p0, const _Float16* p1) {
  FragU f;
  f.u[0] = *(const v4u*)p0;
  f.u[1] = *(const v4u*)p1;
  return f.h;
}

// ---- fused kernel: one workgroup (256 thr = 8 waves) per ray of 128 samples ----
__global__ __launch_bounds__(256) void nerf_fused(
    const float* __restrict__ sigma_feature,
    const float* __restrict__ app_features,
    const float* __restrict__ viewdirs,
    const float* __restrict__ dists,
    const float* __restrict__ z_vals,
    const float* __restrict__ b1,
    const float* __restrict__ b2,
    const float* __restrict__ b3,
    const _Float16* __restrict__ wgt,
    float* __restrict__ out)
{
  extern __shared__ char smem[];
  _Float16* featA = (_Float16*)(smem + FEATA_OFF);
  _Float16* H1    = (_Float16*)(smem + H1_OFF);
  _Float16* H2    = (_Float16*)(smem + H2_OFF);
  float*    wray  = (float*)(smem + WRAY_OFF);
  float*    red   = (float*)(smem + RED_OFF);

  const int tid  = threadIdx.x;
  const int wave = tid >> 5;
  const int lane = tid & 31;
  const int ln16 = lane & 15;
  const int hs   = (lane >> 4) & 1;     // half-wave select
  const int r    = blockIdx.x;

  const float vdx = viewdirs[r * 3 + 0];
  const float vdy = viewdirs[r * 3 + 1];
  const float vdz = viewdirs[r * 3 + 2];
  const float* appR = app_features + (size_t)r * NS * APPD;

  // ---------------- build full 128x416 f16 A matrix in LDS ----------------
  // raw app features (k in [0,27))
  for (int i = tid; i < NS * APPD; i += 256) {
    int s = i / APPD, c = i - s * APPD;
    featA[s * STRA + c] = (_Float16)appR[s * APPD + c];
  }
  // viewdirs (k in [27,30)) + zero padding (k in [390,416))
  for (int s = tid; s < NS; s += 256) {
    _Float16* rowp = featA + s * STRA;
    rowp[27] = (_Float16)vdx;
    rowp[28] = (_Float16)vdy;
    rowp[29] = (_Float16)vdz;
#pragma unroll
    for (int k = 390; k < 416; ++k) rowp[k] = (_Float16)0.0f;
  }
  // PE(app): per (s,d) one sin+cos, then double-angle recurrence over 6 freqs
  for (int e = tid; e < NS * APPD; e += 256) {
    int s = e / APPD, d = e - s * APPD;
    float x  = appR[s * APPD + d];
    float sn = __sinf(x), cs = __cosf(x);
    _Float16* ps = featA + s * STRA + 30 + d * 6;        // sin block [30,192)
    _Float16* pc = ps + 162;                              // cos block [192,354)
#pragma unroll
    for (int f = 0; f < 6; ++f) {
      ps[f] = (_Float16)sn;
      pc[f] = (_Float16)cs;
      float sn2 = 2.0f * sn * cs;
      float cs2 = 1.0f - 2.0f * sn * sn;                  // cos(2x)
      sn = sn2; cs = cs2;
    }
  }
  // PE(viewdirs): sin block [354,372), cos block [372,390)
  for (int e = tid; e < NS * 3; e += 256) {
    int s = e / 3, d = e - s * 3;
    float x  = (d == 0) ? vdx : ((d == 1) ? vdy : vdz);
    float sn = __sinf(x), cs = __cosf(x);
    _Float16* ps = featA + s * STRA + 354 + d * 6;
    _Float16* pc = ps + 18;
#pragma unroll
    for (int f = 0; f < 6; ++f) {
      ps[f] = (_Float16)sn;
      pc[f] = (_Float16)cs;
      float sn2 = 2.0f * sn * cs;
      float cs2 = 1.0f - 2.0f * sn * sn;
      sn = sn2; cs = cs2;
    }
  }
  // density path: softplus -> alpha (alpha[last] = 1)
  if (tid < NS) {
    float x  = sigma_feature[(size_t)r * NS + tid] - 10.0f;
    float sg = (x > 20.0f) ? x : log1pf(__expf(x));
    float a  = 1.0f - __expf(-sg * dists[(size_t)r * NS + tid] * 25.0f);
    if (tid == NS - 1) a = 1.0f;
    wray[tid] = a;
  }
  __syncthreads();
  if (tid == 0) {          // serial cumprod -> weights (off the FLOP critical path)
    float T = 1.0f;
    for (int s = 0; s < NS; ++s) {
      float a = wray[s];
      wray[s] = a * T;
      T *= (1.0f - a + 1e-10f);
    }
  }

  const int row = wave * 16 + ln16;     // A-fragment row for this lane
  const int klo = hs * 8;               // A-fragment K sub-offset

  // ================= GEMM1: [128 x 416] x [416 x 128] ===================
  v8f acc1[8] = {};
  const _Float16* W1h = wgt + W1H_OFF;
  for (int c = 0; c < 13; ++c) {
    const _Float16* ap = featA + row * STRA + c * 32 + klo;
    v16h af = load_frag(ap, ap + 16);
    FragU bf[8];
#pragma unroll
    for (int n = 0; n < 8; ++n) {       // batch all B loads -> one clause
      const _Float16* bp = W1h + (size_t)(n * 16 + ln16) * KP1 + c * 32 + hs * 8;
      bf[n].u[0] = *(const v4u*)bp;
      bf[n].u[1] = *(const v4u*)(bp + 16);
    }
#pragma unroll
    for (int n = 0; n < 8; ++n) {
      acc1[n] = __builtin_amdgcn_wmma_f32_16x16x32_f16(
          false, af, false, bf[n].h, (short)0, acc1[n], false, false);
    }
  }
  __syncthreads();   // featA dead -> H1/H2 may overlay it

  // epilogue: bias + ReLU -> H1 (each wave writes only its own 16-row strip)
#pragma unroll
  for (int n = 0; n < 8; ++n) {
    int col = n * 16 + ln16;
    float bb = b1[col];
#pragma unroll
    for (int g = 0; g < 8; ++g) {
      int srow = wave * 16 + g + hs * 8;
      float h = acc1[n][g] + bb;
      H1[srow * STRH + col] = (_Float16)(h > 0.0f ? h : 0.0f);
    }
  }

  // ================= GEMM2: [128 x 128] x [128 x 128] ===================
  v8f acc2[8] = {};
  const _Float16* W2h = wgt + W2H_OFF;
  for (int c = 0; c < 4; ++c) {
    const _Float16* ap = H1 + row * STRH + c * 32 + klo;
    v16h af = load_frag(ap, ap + 16);
    FragU bf[8];
#pragma unroll
    for (int n = 0; n < 8; ++n) {
      const _Float16* bp = W2h + (size_t)(n * 16 + ln16) * FC + c * 32 + hs * 8;
      bf[n].u[0] = *(const v4u*)bp;
      bf[n].u[1] = *(const v4u*)(bp + 16);
    }
#pragma unroll
    for (int n = 0; n < 8; ++n) {
      acc2[n] = __builtin_amdgcn_wmma_f32_16x16x32_f16(
          false, af, false, bf[n].h, (short)0, acc2[n], false, false);
    }
  }
#pragma unroll
  for (int n = 0; n < 8; ++n) {
    int col = n * 16 + ln16;
    float bb = b2[col];
#pragma unroll
    for (int g = 0; g < 8; ++g) {
      int srow = wave * 16 + g + hs * 8;
      float h = acc2[n][g] + bb;
      H2[srow * STRH + col] = (_Float16)(h > 0.0f ? h : 0.0f);
    }
  }

  // ================= GEMM3: [128 x 128] x [128 x 16(3)] =================
  v8f acc3 = {};
  const _Float16* W3h = wgt + W3H_OFF;
#pragma unroll
  for (int c = 0; c < 4; ++c) {
    const _Float16* ap = H2 + row * STRH + c * 32 + klo;
    v16h af = load_frag(ap, ap + 16);
    const _Float16* bp = W3h + (size_t)ln16 * FC + c * 32 + hs * 8;
    v16h bf = load_frag(bp, bp + 16);
    acc3 = __builtin_amdgcn_wmma_f32_16x16x32_f16(
        false, af, false, bf, (short)0, acc3, false, false);
  }
  // sigmoid + pre-multiply by compositing weights -> LDS
  if (ln16 < 3) {
    float bb = b3[ln16];
#pragma unroll
    for (int g = 0; g < 8; ++g) {
      int s = wave * 16 + g + hs * 8;
      float o  = acc3[g] + bb;
      float sg = 1.0f / (1.0f + __expf(-o));
      red[s * 4 + ln16] = sg * wray[s];
    }
  }
  __syncthreads();

  // ================= per-ray composite reduction =========================
  if (tid < 32) {
    float pr = 0.f, pg = 0.f, pb = 0.f, pd = 0.f, pa = 0.f;
    for (int s = tid; s < NS; s += 32) {
      float w = wray[s];
      pr += red[s * 4 + 0];
      pg += red[s * 4 + 1];
      pb += red[s * 4 + 2];
      pd += w * z_vals[(size_t)r * NS + s];
      pa += w;
    }
#pragma unroll
    for (int off = 16; off > 0; off >>= 1) {
      pr += __shfl_xor(pr, off, 32);
      pg += __shfl_xor(pg, off, 32);
      pb += __shfl_xor(pb, off, 32);
      pd += __shfl_xor(pd, off, 32);
      pa += __shfl_xor(pa, off, 32);
    }
    if (tid == 0) {
      out[r * 3 + 0] = pr;
      out[r * 3 + 1] = pg;
      out[r * 3 + 2] = pb;
      out[N_RAYS * 3 + r] = pd;   // depth_map
      out[N_RAYS * 4 + r] = pa;   // acc_map
    }
  }
}

extern "C" void kernel_launch(void* const* d_in, const int* in_sizes, int n_in,
                              void* d_out, int out_size, void* d_ws, size_t ws_size,
                              hipStream_t stream) {
  (void)in_sizes; (void)n_in; (void)out_size; (void)ws_size;
  const float* sigma = (const float*)d_in[0];
  const float* app   = (const float*)d_in[1];
  const float* vd    = (const float*)d_in[2];
  const float* dists = (const float*)d_in[3];
  const float* zv    = (const float*)d_in[4];
  const float* W1    = (const float*)d_in[5];
  const float* b1    = (const float*)d_in[6];
  const float* W2    = (const float*)d_in[7];
  const float* b2    = (const float*)d_in[8];
  const float* W3    = (const float*)d_in[9];
  const float* b3    = (const float*)d_in[10];
  _Float16* wgt = (_Float16*)d_ws;
  float* out = (float*)d_out;

  // allow >64KB dynamic LDS (CDNA5 WGP has 320KB)
  (void)hipFuncSetAttribute((const void*)nerf_fused,
                            hipFuncAttributeMaxDynamicSharedMemorySize, SMEM_TOT);

  prep_weights<<<(WTOT + 255) / 256, 256, 0, stream>>>(W1, W2, W3, wgt);
  nerf_fused<<<N_RAYS, 256, SMEM_TOT, stream>>>(
      sigma, app, vd, dists, zv, b1, b2, b3, wgt, out);
}
